// VectorQuantizerEMA_89653147337033
// MI455X (gfx1250) — matrix operand order repaired and checked by reading, not compiled, not run on hardware
//
#include <hip/hip_runtime.h>
#include <hip/hip_bf16.h>
#include <math.h>

// ---------------------------------------------------------------------------
// VQ-VAE EMA quantizer for MI455X (gfx1250, wave32, WMMA).
// Distance GEMM (65536x1024, D=256) done with split-bf16 (hi/lo) WMMA:
// dot(f32) ~= hi*hi + hi*lo + lo*hi accumulated in f32 via
// v_wmma_f32_16x16x32_bf16, with THREE independent accumulator chains so
// back-to-back WMMAs are never RAW-dependent (avoids XDL stall chain).
// ---------------------------------------------------------------------------

typedef __attribute__((ext_vector_type(16))) __bf16 v16bf;
typedef __attribute__((ext_vector_type(8)))  float  v8f;

#define DECAY_F  0.99f
#define OMD_F    0.01f
#define EPS_F    1e-5f

// d_out offsets (floats), outputs concatenated in reference return order
#define OFF_OUT   0            // 4*256*16*32*32 = 16777216
#define OFF_LOSS  16777216     // 1
#define OFF_INDS  16777217     // 65536
#define OFF_PERP  16842753     // 1
#define OFF_NEMB  16842754     // 262144
#define OFF_NCS   17104898     // 1024
#define OFF_NEA   17105922     // 262144

// workspace layout (bytes)
#define WS_EMB_HI   0u          // 1024*256 ushort
#define WS_EMB_LO   524288u     // 1024*256 ushort
#define WS_COLNORM  1048576u    // 1024 f32
#define WS_COUNTS   1052672u    // 1024 f32
#define WS_DISTSUM  1056768u    // 1 f32 (padded)
#define WS_INDS     1057024u    // 65536 int
#define WS_EMBSUM   1319168u    // 1024*256 f32

static __device__ __forceinline__ unsigned short f32_to_bf16(float f) {
    unsigned u = __float_as_uint(f);
    u += 0x7FFFu + ((u >> 16) & 1u);   // round-to-nearest-even
    return (unsigned short)(u >> 16);
}
static __device__ __forceinline__ float bf16_to_f32(unsigned short h) {
    return __uint_as_float(((unsigned)h) << 16);
}

// ---------------------------------------------------------------------------
// Kernel 1: embedding -> bf16 hi/lo, exact col norms, zero accumulators.
// grid = 1024 blocks (one code row), 256 threads (one per channel).
// ---------------------------------------------------------------------------
__global__ void vq_prep(const float* __restrict__ emb,
                        unsigned short* __restrict__ embHi,
                        unsigned short* __restrict__ embLo,
                        float* __restrict__ colNorm,
                        float* __restrict__ counts,
                        float* __restrict__ embSum,
                        float* __restrict__ distSum) {
    __shared__ float red[256];
    const int k = blockIdx.x, d = threadIdx.x;
    const int i = (k << 8) + d;
    float x = emb[i];
    unsigned short hi = f32_to_bf16(x);
    unsigned short lo = f32_to_bf16(x - bf16_to_f32(hi));
    embHi[i] = hi;
    embLo[i] = lo;
    embSum[i] = 0.0f;
    red[d] = x * x;
    __syncthreads();
    for (int s = 128; s > 0; s >>= 1) {
        if (d < s) red[d] += red[d + s];
        __syncthreads();
    }
    if (d == 0) {
        colNorm[k] = red[0];
        counts[k]  = 0.0f;
        if (k == 0) distSum[0] = 0.0f;
    }
}

// ---------------------------------------------------------------------------
// Kernel 2: distances + argmin via WMMA.
// grid = 1024 blocks x 128 threads (4 waves). Block handles 64 latent rows
// from one (b,z) slab. LDS holds A-fragments (hi/lo) in contiguous-k layout (k = 16*half + slot, as B);
// compiler promotes them to VGPRs across the K loop.
// ---------------------------------------------------------------------------
__global__ void __launch_bounds__(128)
vq_argmin(const float* __restrict__ latents,
          const unsigned short* __restrict__ embHi,
          const unsigned short* __restrict__ embLo,
          const float* __restrict__ colNorm,
          int*   __restrict__ indsWs,
          float* __restrict__ indsOutF,
          float* __restrict__ counts,
          float* __restrict__ distSum) {
    extern __shared__ unsigned char smem[];
    unsigned short* AhiL   = (unsigned short*)smem;            // 32768 B
    unsigned short* AloL   = AhiL + 16384;                     // 32768 B
    float*          partial = (float*)(smem + 65536);          // 512 B
    float*          rowNorm = partial + 128;                   // 256 B

    const int tid = threadIdx.x;
    const int n0  = blockIdx.x * 64;
    const int slab = n0 >> 10;               // (b,z) slab: 1024 rows each
    const int bI = slab >> 4, zI = slab & 15;
    const int hw0 = n0 & 1023;
    const size_t latBase = (size_t)bI * 4194304u + (size_t)zI * 1024u + (size_t)hw0;

    // ---- stage A (64 rows x 256 dims) into WMMA fragment layout, bf16 hi/lo
    // A layout used here (contiguous k, as B): lane = m + 16*bit4(k); per-lane bf16 slot
    // j = k&15
    float acc = 0.0f;
    const int rr = tid & 63;
    for (int s = 0; s < 128; ++s) {
        const int d = (tid >> 6) + 2 * s;
        const float x = latents[latBase + (size_t)d * 16384u + (size_t)rr];
        acc += x * x;
        const unsigned short hi = f32_to_bf16(x);
        const unsigned short lo = f32_to_bf16(x - bf16_to_f32(hi));
        const int w = rr >> 4, m = rr & 15, dc = d >> 5, kk = d & 31;
        const int lane = m + (((kk >> 4) & 1) << 4);
        const int j = kk & 15;
        const int idx = (((w << 3) + dc) * 32 + lane) * 16 + j;
        AhiL[idx] = hi;
        AloL[idx] = lo;
    }
    partial[tid] = acc;
    __syncthreads();
    if (tid < 64) rowNorm[tid] = partial[tid] + partial[tid + 64];
    __syncthreads();

    // ---- per-wave 16x16 output tile, loop over 64 K-tiles
    const int wv   = tid >> 5;
    const int lane = tid & 31;
    const int half = lane >> 4;
    const int colL = lane & 15;
    const int mbase = (wv << 4) + (half << 3);   // D layout: VGPR i -> M=i(+8)

    float rn[8];
    #pragma unroll
    for (int i = 0; i < 8; ++i) rn[i] = rowNorm[mbase + i];

    float best[8];
    int   bidx[8];
    #pragma unroll
    for (int i = 0; i < 8; ++i) { best[i] = 3.4e38f; bidx[i] = 0; }

    for (int kt = 0; kt < 64; ++kt) {
        const int code = (kt << 4) + colL;
        const unsigned short* bhRow = embHi + (code << 8) + (half << 4);
        const unsigned short* blRow = embLo + (code << 8) + (half << 4);
        if (kt + 1 < 64) {
            __builtin_prefetch(bhRow + (16 << 8), 0, 1);   // global_prefetch_b8
            __builtin_prefetch(blRow + (16 << 8), 0, 1);
        }
        // three independent accumulator chains: hi*hi, hi*lo, lo*hi
        v8f c1 = {0.f, 0.f, 0.f, 0.f, 0.f, 0.f, 0.f, 0.f};
        v8f c2 = {0.f, 0.f, 0.f, 0.f, 0.f, 0.f, 0.f, 0.f};
        v8f c3 = {0.f, 0.f, 0.f, 0.f, 0.f, 0.f, 0.f, 0.f};
        #pragma unroll
        for (int dc = 0; dc < 8; ++dc) {
            const v16bf bh = *(const v16bf*)(bhRow + dc * 32);
            const v16bf bl = *(const v16bf*)(blRow + dc * 32);
            const int fo = (((wv << 3) + dc) * 32 + lane) * 16;
            const v16bf ah = *(const v16bf*)(AhiL + fo);
            const v16bf al = *(const v16bf*)(AloL + fo);
            c1 = __builtin_amdgcn_wmma_f32_16x16x32_bf16(false, ah, false, bh, (short)0, c1, false, false);
            c2 = __builtin_amdgcn_wmma_f32_16x16x32_bf16(false, ah, false, bl, (short)0, c2, false, false);
            c3 = __builtin_amdgcn_wmma_f32_16x16x32_bf16(false, al, false, bh, (short)0, c3, false, false);
        }
        const float cn = colNorm[code];
        #pragma unroll
        for (int i = 0; i < 8; ++i) {
            const float dot = c1[i] + c2[i] + c3[i];
            const float dist = rn[i] + cn - 2.0f * dot;
            if (dist < best[i]) { best[i] = dist; bidx[i] = code; }
        }
    }

    // ---- butterfly argmin across the 16 lanes sharing each row (half kept)
    #pragma unroll
    for (int mask = 1; mask <= 8; mask <<= 1) {
        #pragma unroll
        for (int i = 0; i < 8; ++i) {
            const float od = __shfl_xor(best[i], mask, 32);
            const int   oi = __shfl_xor(bidx[i], mask, 32);
            if (od < best[i] || (od == best[i] && oi < bidx[i])) {
                best[i] = od; bidx[i] = oi;
            }
        }
    }

    if (colL == 0) {
        float ds = 0.0f;
        #pragma unroll
        for (int i = 0; i < 8; ++i) {
            const int n = n0 + mbase + i;
            indsWs[n]   = bidx[i];
            indsOutF[n] = (float)bidx[i];
            atomicAdd(&counts[bidx[i]], 1.0f);
            ds += best[i];
        }
        atomicAdd(distSum, ds);   // sum of min dists == sum ||x - e||^2
    }
}

// ---------------------------------------------------------------------------
// Kernel 3: gather codebook into output layout + segment-sum into embSum.
// grid = 64 slabs * 8 channel-chunks; 256 threads over hw.
// ---------------------------------------------------------------------------
__global__ void vq_scatter(const float* __restrict__ latents,
                           const float* __restrict__ emb,
                           const int* __restrict__ indsWs,
                           float* __restrict__ embSum,
                           float* __restrict__ outp) {
    const int s  = blockIdx.x >> 3;
    const int c0 = (blockIdx.x & 7) << 5;
    const int bI = s >> 4, zI = s & 15;
    const size_t base = (size_t)bI * 4194304u + (size_t)zI * 1024u;
    for (int hw = threadIdx.x; hw < 1024; hw += 256) {
        const int n = (s << 10) + hw;
        const int ind = indsWs[n];
        const float* er = emb + (ind << 8);
        float* es = embSum + (ind << 8);
        #pragma unroll
        for (int cc = 0; cc < 32; ++cc) {
            const int cI = c0 + cc;
            const size_t li = base + (size_t)cI * 16384u + (size_t)hw;
            atomicAdd(&es[cI], latents[li]);
            outp[li] = er[cI];
        }
    }
}

// ---------------------------------------------------------------------------
// Kernel 4: EMA updates, normalized codebook, perplexity, loss scalar.
// single block, 1024 threads (one per code).
// ---------------------------------------------------------------------------
__global__ void __launch_bounds__(1024)
vq_finalize(const float* __restrict__ clusterSize,
            const float* __restrict__ embAvg,
            const float* __restrict__ counts,
            const float* __restrict__ embSum,
            const float* __restrict__ distSum,
            float* __restrict__ dOut) {
    __shared__ float red[1024];
    __shared__ float csL[1024];
    const int t = threadIdx.x;
    const float cnt = counts[t];
    const float ncs = clusterSize[t] * DECAY_F + OMD_F * cnt;
    dOut[OFF_NCS + t] = ncs;

    red[t] = ncs;
    __syncthreads();
    for (int s = 512; s > 0; s >>= 1) {
        if (t < s) red[t] += red[t + s];
        __syncthreads();
    }
    const float ntot = red[0];
    __syncthreads();

    const float avg = cnt * (1.0f / 65536.0f);
    red[t] = avg * __logf(avg + 1e-10f);
    __syncthreads();
    for (int s = 512; s > 0; s >>= 1) {
        if (t < s) red[t] += red[t + s];
        __syncthreads();
    }
    if (t == 0) {
        dOut[OFF_PERP] = __expf(-red[0]);
        dOut[OFF_LOSS] = 1.25f * distSum[0] * (1.0f / 16777216.0f);
    }

    csL[t] = (ncs + EPS_F) / (ntot + 1024.0f * EPS_F) * ntot;
    __syncthreads();

    for (int i = t; i < 1024 * 256; i += 1024) {
        const int k = i >> 8;
        const float nea = embAvg[i] * DECAY_F + OMD_F * embSum[i];
        dOut[OFF_NEA + i]  = nea;
        dOut[OFF_NEMB + i] = nea / csL[k];
    }
}

// ---------------------------------------------------------------------------
extern "C" void kernel_launch(void* const* d_in, const int* in_sizes, int n_in,
                              void* d_out, int out_size, void* d_ws, size_t ws_size,
                              hipStream_t stream) {
    (void)in_sizes; (void)n_in; (void)out_size; (void)ws_size;
    const float* latents     = (const float*)d_in[0];
    const float* embedding   = (const float*)d_in[1];
    const float* clusterSize = (const float*)d_in[2];
    const float* embAvg      = (const float*)d_in[3];
    float* out = (float*)d_out;

    unsigned char* ws = (unsigned char*)d_ws;
    unsigned short* embHi   = (unsigned short*)(ws + WS_EMB_HI);
    unsigned short* embLo   = (unsigned short*)(ws + WS_EMB_LO);
    float*          colNorm = (float*)(ws + WS_COLNORM);
    float*          counts  = (float*)(ws + WS_COUNTS);
    float*          distSum = (float*)(ws + WS_DISTSUM);
    int*            indsWs  = (int*)(ws + WS_INDS);
    float*          embSum  = (float*)(ws + WS_EMBSUM);

    vq_prep<<<1024, 256, 0, stream>>>(embedding, embHi, embLo, colNorm,
                                      counts, embSum, distSum);

    const size_t lds = 64u * 256u * 2u * 2u + 128u * 4u + 64u * 4u; // 66304 B
    vq_argmin<<<1024, 128, lds, stream>>>(latents, embHi, embLo, colNorm,
                                          indsWs, out + OFF_INDS, counts, distSum);

    vq_scatter<<<512, 256, 0, stream>>>(latents, embedding, indsWs, embSum,
                                        out + OFF_OUT);

    vq_finalize<<<1, 1024, 0, stream>>>(clusterSize, embAvg, counts, embSum,
                                        distSum, out);
}
